// RNN_44220983280126
// MI455X (gfx1250) — matrix operand order repaired and checked by reading, not compiled, not run on hardware
//
#include <hip/hip_runtime.h>
#include <hip/hip_bf16.h>
#include <stdint.h>

typedef __attribute__((ext_vector_type(16))) _Float16 v16h;
typedef __attribute__((ext_vector_type(8)))  _Float16 v8h;
typedef __attribute__((ext_vector_type(8)))  float    v8f;
typedef __attribute__((ext_vector_type(4)))  unsigned int v4u;
typedef __attribute__((ext_vector_type(8)))  int      v8i_;
typedef __attribute__((ext_vector_type(4)))  int      v4i_;

#define BQ    256
#define LQ    128
#define EQ    300
#define EPAD  320
#define HIDQ  256
#define GATES 1024
#define D1Q   100
#define D1PAD 128
#define CQ    4
#define NBLK_LSTM 32

// ---------------------------------------------------------------------------
// WMMA fragment loaders (gfx1250 wave32 16x16x32 f16 layouts, ISA 7.12.2)
// ---------------------------------------------------------------------------
__device__ __forceinline__ v16h load_a_frag(const _Float16* __restrict__ tile,
                                            int ld, int k0, int lane) {
  int r  = lane & 15;
  int kb = (lane >> 4) << 3;
  const _Float16* p = tile + (size_t)r * ld + (k0 + kb);
  v8h lo = *(const v8h*)(p);
  v8h hi = *(const v8h*)(p + 16);
  v16h f;
#pragma unroll
  for (int i = 0; i < 8; ++i) { f[i] = lo[i]; f[i + 8] = hi[i]; }
  return f;
}

__device__ __forceinline__ v16h load_b_frag(const _Float16* __restrict__ tileT,
                                            int ld, int k0, int lane) {
  int n  = lane & 15;
  int kb = (lane >> 4) << 3;
  const _Float16* p = tileT + (size_t)n * ld + (k0 + kb);
  v8h lo = *(const v8h*)(p);
  v8h hi = *(const v8h*)(p + 16);
  v16h f;
#pragma unroll
  for (int i = 0; i < 8; ++i) { f[i] = lo[i]; f[i + 8] = hi[i]; }
  return f;
}

// ---------------------------------------------------------------------------
// Tensor Data Mover: 2D tile (rows x cols f16, row stride in elems) -> LDS.
// D# built per CDNA5 ISA 8.3-8.4. Hedged across 5-arg (ROCm7.2) / 6-arg
// (therock) builtin arities via the TDM header presence.
// ---------------------------------------------------------------------------
#if defined(__has_builtin)
#if __has_builtin(__builtin_amdgcn_tensor_load_to_lds)
#define HAVE_TDM 1
#endif
#endif

#ifdef HAVE_TDM
__device__ __forceinline__ void tdm_load_2d(unsigned lds_off, const void* gsrc,
                                            unsigned rows, unsigned cols,
                                            unsigned stride_elems) {
  unsigned long long ga = (unsigned long long)(uintptr_t)gsrc;
  v4u g0;
  g0[0] = 1u;                                            // count=1, user mode
  g0[1] = lds_off;                                       // lds_addr
  g0[2] = (unsigned)(ga & 0xFFFFFFFFu);                  // global_addr lo
  g0[3] = (unsigned)((ga >> 32) & 0x01FFFFFFu) | 0x80000000u; // addr hi | type=2
  v8i_ g1;
  g1[0] = 0x00010000;                                    // data_size=1 (2 bytes)
  g1[1] = (int)((cols & 0xFFFFu) << 16);                 // tensor_dim0 lo16
  g1[2] = (int)(((cols >> 16) & 0xFFFFu) | ((rows & 0xFFFFu) << 16)); // d0 hi | d1 lo
  g1[3] = (int)(((rows >> 16) & 0xFFFFu) | ((cols & 0xFFFFu) << 16)); // d1 hi | tile_dim0
  g1[4] = (int)(rows & 0xFFFFu);                         // tile_dim1 (tile_dim2=0)
  g1[5] = (int)stride_elems;                             // tensor_dim0_stride lo32
  g1[6] = 0;
  g1[7] = 0;
  v4i_ gz = {0, 0, 0, 0};
#if __has_include(<hip/amd_detail/amd_gfx1250_TDM.h>)
  v8i_ gz8 = {0, 0, 0, 0, 0, 0, 0, 0};
  __builtin_amdgcn_tensor_load_to_lds(g0, g1, gz, gz, gz8, 0);
#else
  __builtin_amdgcn_tensor_load_to_lds(g0, g1, gz, gz, 0);
#endif
}
#endif

// ---------------------------------------------------------------------------
// Generic batched WMMA GEMM (global-fed): C = alpha * A[M,K] * (Bt[N,K])^T + bias
// SMODE: 0 f32 out, 1 f16 out, 2 f16 out transposed per 128-row batch (V^T)
// ---------------------------------------------------------------------------
template <int SMODE, bool RELU>
__global__ void __launch_bounds__(256)
wmma_gemm_kernel(const _Float16* __restrict__ A, const _Float16* __restrict__ Bt,
                 const float* __restrict__ bias, void* __restrict__ Cout,
                 int K, int lda, int ldb, int ldc,
                 long long sA, long long sB, long long sC,
                 int ntiles4, int tilesPerBatch, int totalTiles, float alpha) {
  int wave = blockIdx.x * 8 + (threadIdx.x >> 5);
  if (wave >= totalTiles) return;
  int batch = wave / tilesPerBatch;
  int rem   = wave - batch * tilesPerBatch;
  int mt = rem / ntiles4;
  int nt = rem - mt * ntiles4;
  int lane = threadIdx.x & 31;

  const _Float16* Atile = A  + (size_t)batch * sA + (size_t)(mt * 16) * lda;
  const _Float16* Btile = Bt + (size_t)batch * sB + (size_t)(nt * 64) * ldb;

  v8f acc0 = {}, acc1 = {}, acc2 = {}, acc3 = {};
  for (int k0 = 0; k0 < K; k0 += 32) {
    v16h a  = load_a_frag(Atile, lda, k0, lane);
    v16h b0 = load_b_frag(Btile,                    ldb, k0, lane);
    v16h b1 = load_b_frag(Btile + (size_t)16 * ldb, ldb, k0, lane);
    v16h b2 = load_b_frag(Btile + (size_t)32 * ldb, ldb, k0, lane);
    v16h b3 = load_b_frag(Btile + (size_t)48 * ldb, ldb, k0, lane);
    acc0 = __builtin_amdgcn_wmma_f32_16x16x32_f16(false, a, false, b0, (short)0, acc0, false, false);
    acc1 = __builtin_amdgcn_wmma_f32_16x16x32_f16(false, a, false, b1, (short)0, acc1, false, false);
    acc2 = __builtin_amdgcn_wmma_f32_16x16x32_f16(false, a, false, b2, (short)0, acc2, false, false);
    acc3 = __builtin_amdgcn_wmma_f32_16x16x32_f16(false, a, false, b3, (short)0, acc3, false, false);
  }

  int nb    = nt * 64 + (lane & 15);
  int mbase = mt * 16 + ((lane >> 4) << 3);
  v8f accs[4] = {acc0, acc1, acc2, acc3};
#pragma unroll
  for (int j = 0; j < 4; ++j) {
    int n = nb + j * 16;
    float bval = bias ? bias[n] : 0.0f;
#pragma unroll
    for (int r = 0; r < 8; ++r) {
      float vout = accs[j][r] * alpha + bval;
      if (RELU) vout = fmaxf(vout, 0.0f);
      int m = mbase + r;
      if (SMODE == 0) {
        ((float*)Cout)[(size_t)batch * sC + (size_t)m * ldc + n] = vout;
      } else if (SMODE == 1) {
        ((_Float16*)Cout)[(size_t)batch * sC + (size_t)m * ldc + n] = (_Float16)vout;
      } else {
        int bb = m >> 7, l = m & (LQ - 1);
        ((_Float16*)Cout)[((size_t)bb * HIDQ + n) * LQ + l] = (_Float16)vout;
      }
    }
  }
}

// ---------------------------------------------------------------------------
// Persistent LSTM kernel: all 128 timesteps in one launch.
// 32 WGs x 256 threads. WG g: batch-slice bs=g>>2 (32 rows), j-slice js=g&3
// (64 hidden units, all 4 gates). Wave w: gate=w&3, mtile=w>>2.
// Per step: TDM-stage h tile to LDS -> WMMA (A from LDS, B from L2-hot WhT)
// -> zh exchange via LDS -> fused gate math -> grid barrier (global atomic).
// h ping-pongs between hA/hB; c is WG-exclusive so needs no sync.
// ---------------------------------------------------------------------------
__device__ __forceinline__ float sigmoidf_(float x) { return 1.0f / (1.0f + __expf(-x)); }

__global__ void __launch_bounds__(256)
lstm_persistent_kernel(const _Float16* __restrict__ gatesx,  // [B*L,1024], bias folded
                       const _Float16* __restrict__ WhT,     // [1024][256] f16
                       _Float16* __restrict__ hA,            // [256][256] f16, zeroed
                       _Float16* __restrict__ hB,            // [256][256] f16, zeroed
                       float* __restrict__ cG,               // [256][256] f32, zeroed
                       _Float16* __restrict__ hseq,          // [B*L][256]
                       unsigned int* __restrict__ bar) {     // zeroed
  __shared__ __align__(32) _Float16 sAh[32 * HIDQ];   // 16 KB staged h tile
  __shared__ __align__(16) float    sZ[32 * HIDQ];    // 32 KB zh exchange

  const int g    = blockIdx.x;
  const int bs   = g >> 2;            // [0,8)
  const int js   = g & 3;             // [0,4)
  const int w    = threadIdx.x >> 5;  // wave in WG
  const int lane = threadIdx.x & 31;
  const int gate = w & 3;
  const int ml   = w >> 2;            // [0,2)
  const int nt   = gate * 4 + js;     // global 64-wide n tile

  const _Float16* Btile = WhT + (size_t)(nt * 64) * HIDQ;

  for (int t = 0; t < LQ; ++t) {
    const _Float16* hcur = (t & 1) ? hB : hA;
    _Float16*       hnxt = (t & 1) ? hA : hB;

    // ---- stage A = h[bs*32 .. +32][0..256) into LDS ----
#ifdef HAVE_TDM
    if (w == 0) {
      tdm_load_2d((unsigned)(uintptr_t)(void*)sAh,
                  hcur + (size_t)(bs * 32) * HIDQ, 32u, (unsigned)HIDQ, (unsigned)HIDQ);
      __builtin_amdgcn_s_wait_tensorcnt(0);
    }
#else
    {
      const v8h* src = (const v8h*)(hcur + (size_t)(bs * 32) * HIDQ);
      v8h* dst = (v8h*)sAh;
      for (int i = threadIdx.x; i < 32 * HIDQ / 8; i += 256) dst[i] = src[i];
    }
#endif
    __syncthreads();

    // ---- zh tile: 16x64, A from LDS, B from global ----
    const _Float16* Als = sAh + (size_t)(ml * 16) * HIDQ;
    v8f acc0 = {}, acc1 = {}, acc2 = {}, acc3 = {};
#pragma unroll
    for (int k0 = 0; k0 < HIDQ; k0 += 32) {
      v16h a  = load_a_frag(Als, HIDQ, k0, lane);
      v16h b0 = load_b_frag(Btile,                      HIDQ, k0, lane);
      v16h b1 = load_b_frag(Btile + (size_t)16 * HIDQ,  HIDQ, k0, lane);
      v16h b2 = load_b_frag(Btile + (size_t)32 * HIDQ,  HIDQ, k0, lane);
      v16h b3 = load_b_frag(Btile + (size_t)48 * HIDQ,  HIDQ, k0, lane);
      acc0 = __builtin_amdgcn_wmma_f32_16x16x32_f16(false, a, false, b0, (short)0, acc0, false, false);
      acc1 = __builtin_amdgcn_wmma_f32_16x16x32_f16(false, a, false, b1, (short)0, acc1, false, false);
      acc2 = __builtin_amdgcn_wmma_f32_16x16x32_f16(false, a, false, b2, (short)0, acc2, false, false);
      acc3 = __builtin_amdgcn_wmma_f32_16x16x32_f16(false, a, false, b3, (short)0, acc3, false, false);
    }
    // stash zh into LDS: sZ[b_local(32)][gate*64 + col]
    {
      int col0  = lane & 15;
      int mbase = ml * 16 + ((lane >> 4) << 3);
      v8f accs[4] = {acc0, acc1, acc2, acc3};
#pragma unroll
      for (int j = 0; j < 4; ++j) {
        int col = j * 16 + col0;
#pragma unroll
        for (int r = 0; r < 8; ++r)
          sZ[(size_t)(mbase + r) * HIDQ + gate * 64 + col] = accs[j][r];
      }
    }
    __syncthreads();

    // ---- fused gate math: 32 x 64 (b_local, col) pairs, 8 per thread ----
#pragma unroll
    for (int p = 0; p < 8; ++p) {
      int e   = threadIdx.x + p * 256;      // [0,2048)
      int bl  = e >> 6;                     // [0,32)
      int col = e & 63;                     // [0,64)
      int b   = bs * 32 + bl;
      int j   = js * 64 + col;
      size_t grow = (size_t)(b * LQ + t) * GATES;
      float zi = sZ[(size_t)bl * HIDQ +   0 + col] + (float)gatesx[grow +          j];
      float zf = sZ[(size_t)bl * HIDQ +  64 + col] + (float)gatesx[grow + HIDQ   + j];
      float zg = sZ[(size_t)bl * HIDQ + 128 + col] + (float)gatesx[grow + 2*HIDQ + j];
      float zo = sZ[(size_t)bl * HIDQ + 192 + col] + (float)gatesx[grow + 3*HIDQ + j];
      size_t cidx = (size_t)b * HIDQ + j;
      float cn = sigmoidf_(zf) * cG[cidx] + sigmoidf_(zi) * tanhf(zg);
      cG[cidx] = cn;
      float h = sigmoidf_(zo) * tanhf(cn);
      hnxt[cidx] = (_Float16)h;
      hseq[(size_t)(b * LQ + t) * HIDQ + j] = (_Float16)h;
    }

    // ---- grid barrier (monotonic counter, counter zeroed per launch) ----
    __threadfence();
    __syncthreads();
    if (threadIdx.x == 0) {
      atomicAdd(bar, 1u);
      unsigned target = (unsigned)(t + 1) * NBLK_LSTM;
      while (__hip_atomic_load(bar, __ATOMIC_ACQUIRE, __HIP_MEMORY_SCOPE_AGENT) < target)
        __builtin_amdgcn_s_sleep(1);
    }
    __syncthreads();
  }
}

// ---------------------------------------------------------------------------
// Helper / elementwise kernels
// ---------------------------------------------------------------------------
__global__ void embed_cast_kernel(const int* __restrict__ x,
                                  const float* __restrict__ embed,
                                  _Float16* __restrict__ out) {
  size_t i = (size_t)blockIdx.x * blockDim.x + threadIdx.x;
  size_t total = (size_t)BQ * LQ * EPAD;
  if (i >= total) return;
  int row = (int)(i / EPAD);
  int kk  = (int)(i - (size_t)row * EPAD);
  out[i] = (kk < EQ) ? (_Float16)embed[(size_t)x[row] * EQ + kk] : (_Float16)0.0f;
}

__global__ void transpose_cast_kernel(const float* __restrict__ W,
                                      _Float16* __restrict__ Wt,
                                      int K, int N, int Kpad, int Npad) {
  size_t i = (size_t)blockIdx.x * blockDim.x + threadIdx.x;
  size_t total = (size_t)Npad * Kpad;
  if (i >= total) return;
  int n = (int)(i / Kpad);
  int k = (int)(i - (size_t)n * Kpad);
  Wt[i] = (n < N && k < K) ? (_Float16)W[(size_t)k * N + n] : (_Float16)0.0f;
}

__global__ void pad_bias_kernel(const float* __restrict__ b, float* __restrict__ bp,
                                int N, int Npad) {
  int i = blockIdx.x * blockDim.x + threadIdx.x;
  if (i < Npad) bp[i] = (i < N) ? b[i] : 0.0f;
}

__global__ void init_state_kernel(_Float16* __restrict__ hA, _Float16* __restrict__ hB,
                                  float* __restrict__ c, unsigned int* __restrict__ bar) {
  int i = blockIdx.x * blockDim.x + threadIdx.x;
  if (i < BQ * HIDQ) { hA[i] = (_Float16)0.0f; hB[i] = (_Float16)0.0f; c[i] = 0.0f; }
  if (i == 0) *bar = 0u;
}

__global__ void __launch_bounds__(256)
attn_softmax_kernel(const float* __restrict__ scores,
                    const unsigned char* __restrict__ mask,
                    _Float16* __restrict__ attn) {
  int wave = blockIdx.x * 8 + (threadIdx.x >> 5);
  if (wave >= BQ * LQ) return;
  int lane = threadIdx.x & 31;
  int b = wave >> 7, qi = wave & (LQ - 1);
  bool valid = mask[b * LQ + qi] != 0;
  const float* row = scores + (size_t)wave * LQ;
  float v[4];
#pragma unroll
  for (int i = 0; i < 4; ++i) v[i] = valid ? row[lane + 32 * i] : 0.0f;
  float mx = fmaxf(fmaxf(v[0], v[1]), fmaxf(v[2], v[3]));
#pragma unroll
  for (int off = 16; off > 0; off >>= 1) mx = fmaxf(mx, __shfl_xor(mx, off, 32));
  float e[4], s = 0.0f;
#pragma unroll
  for (int i = 0; i < 4; ++i) { e[i] = __expf(v[i] - mx); s += e[i]; }
#pragma unroll
  for (int off = 16; off > 0; off >>= 1) s += __shfl_xor(s, off, 32);
  float inv = 1.0f / s;
  _Float16* orow = attn + (size_t)wave * LQ;
#pragma unroll
  for (int i = 0; i < 4; ++i) orow[lane + 32 * i] = (_Float16)(e[i] * inv);
}

__global__ void mlp2_softmax_kernel(const float* __restrict__ relu1,
                                    const float* __restrict__ W2,
                                    const float* __restrict__ b2,
                                    float* __restrict__ out) {
  int row = blockIdx.x * blockDim.x + threadIdx.x;
  if (row >= BQ) return;
  float acc[CQ];
#pragma unroll
  for (int c = 0; c < CQ; ++c) acc[c] = b2[c];
  for (int k = 0; k < D1Q; ++k) {
    float a = relu1[(size_t)row * D1PAD + k];
#pragma unroll
    for (int c = 0; c < CQ; ++c) acc[c] += a * W2[k * CQ + c];
  }
  float mx = acc[0];
#pragma unroll
  for (int c = 1; c < CQ; ++c) mx = fmaxf(mx, acc[c]);
  float s = 0.0f, e[CQ];
#pragma unroll
  for (int c = 0; c < CQ; ++c) { e[c] = __expf(acc[c] - mx); s += e[c]; }
  float inv = 1.0f / s;
#pragma unroll
  for (int c = 0; c < CQ; ++c) out[row * CQ + c] = e[c] * inv;
}

// ---------------------------------------------------------------------------
// Host side
// ---------------------------------------------------------------------------
static void run_gemm(hipStream_t stream, int smode, bool relu,
                     const _Float16* A, const _Float16* Bt, const float* bias,
                     void* C, int M, int N, int K, int lda, int ldb, int ldc,
                     long long sA, long long sB, long long sC, int nbatch,
                     float alpha) {
  int ntiles4 = N / 64;
  int tilesPerBatch = (M / 16) * ntiles4;
  int totalTiles = tilesPerBatch * nbatch;
  dim3 g((totalTiles + 7) / 8), b(256);
  if (smode == 0 && !relu)
    wmma_gemm_kernel<0, false><<<g, b, 0, stream>>>(A, Bt, bias, C, K, lda, ldb, ldc, sA, sB, sC, ntiles4, tilesPerBatch, totalTiles, alpha);
  else if (smode == 0)
    wmma_gemm_kernel<0, true><<<g, b, 0, stream>>>(A, Bt, bias, C, K, lda, ldb, ldc, sA, sB, sC, ntiles4, tilesPerBatch, totalTiles, alpha);
  else if (smode == 1)
    wmma_gemm_kernel<1, false><<<g, b, 0, stream>>>(A, Bt, bias, C, K, lda, ldb, ldc, sA, sB, sC, ntiles4, tilesPerBatch, totalTiles, alpha);
  else
    wmma_gemm_kernel<2, false><<<g, b, 0, stream>>>(A, Bt, bias, C, K, lda, ldb, ldc, sA, sB, sC, ntiles4, tilesPerBatch, totalTiles, alpha);
}

extern "C" void kernel_launch(void* const* d_in, const int* in_sizes, int n_in,
                              void* d_out, int out_size, void* d_ws, size_t ws_size,
                              hipStream_t stream) {
  (void)in_sizes; (void)n_in; (void)out_size; (void)ws_size;
  const int*           x      = (const int*)d_in[0];
  const unsigned char* mask   = (const unsigned char*)d_in[1];
  const float* embed  = (const float*)d_in[2];
  const float* Wi     = (const float*)d_in[3];
  const float* Wh     = (const float*)d_in[4];
  const float* b_lstm = (const float*)d_in[5];
  const float* Wq     = (const float*)d_in[6];
  const float* bq     = (const float*)d_in[7];
  const float* Wk     = (const float*)d_in[8];
  const float* bk     = (const float*)d_in[9];
  const float* Wv     = (const float*)d_in[10];
  const float* bv     = (const float*)d_in[11];
  const float* Wo     = (const float*)d_in[12];
  const float* bo     = (const float*)d_in[13];
  const float* W1     = (const float*)d_in[14];
  const float* b1     = (const float*)d_in[15];
  const float* W2     = (const float*)d_in[16];
  const float* b2     = (const float*)d_in[17];

  char* base = (char*)d_ws;
  size_t off = 0;
  auto alloc = [&](size_t bytes) -> void* {
    off = (off + 255) & ~(size_t)255;
    void* p = base + off;
    off += bytes;
    return p;
  };
  _Float16* XembF16 = (_Float16*)alloc((size_t)BQ * LQ * EPAD * 2);
  _Float16* WiT  = (_Float16*)alloc((size_t)GATES * EPAD * 2);
  _Float16* WhT  = (_Float16*)alloc((size_t)GATES * HIDQ * 2);
  _Float16* WqT  = (_Float16*)alloc((size_t)HIDQ * HIDQ * 2);
  _Float16* WkT  = (_Float16*)alloc((size_t)HIDQ * HIDQ * 2);
  _Float16* WvT  = (_Float16*)alloc((size_t)HIDQ * HIDQ * 2);
  _Float16* WoT  = (_Float16*)alloc((size_t)HIDQ * HIDQ * 2);
  _Float16* W1T  = (_Float16*)alloc((size_t)D1PAD * LQ * HIDQ * 2);
  float*    b1p  = (float*)alloc(D1PAD * 4);
  _Float16* gatesx = (_Float16*)alloc((size_t)BQ * LQ * GATES * 2);
  _Float16* hseq   = (_Float16*)alloc((size_t)BQ * LQ * HIDQ * 2);
  _Float16* hPing  = (_Float16*)alloc((size_t)BQ * HIDQ * 2);
  _Float16* hPong  = (_Float16*)alloc((size_t)BQ * HIDQ * 2);
  float*    cSt    = (float*)alloc((size_t)BQ * HIDQ * 4);
  unsigned int* barCnt = (unsigned int*)alloc(256);
  _Float16* qbuf   = (_Float16*)alloc((size_t)BQ * LQ * HIDQ * 2);
  _Float16* kbuf   = (_Float16*)alloc((size_t)BQ * LQ * HIDQ * 2);
  _Float16* vT     = (_Float16*)alloc((size_t)BQ * HIDQ * LQ * 2);
  float*    scores = (float*)alloc((size_t)BQ * LQ * LQ * 4);
  _Float16* attn   = (_Float16*)alloc((size_t)BQ * LQ * LQ * 2);
  _Float16* ctx    = (_Float16*)alloc((size_t)BQ * LQ * HIDQ * 2);
  _Float16* flat   = (_Float16*)alloc((size_t)BQ * LQ * HIDQ * 2);
  float*    relu1  = (float*)alloc((size_t)BQ * D1PAD * 4);

  auto cdiv = [](long long a, long long b) { return (int)((a + b - 1) / b); };

  // ---- weight prep ----
  transpose_cast_kernel<<<cdiv((long long)GATES * EPAD, 256), 256, 0, stream>>>(Wi, WiT, EQ, GATES, EPAD, GATES);
  transpose_cast_kernel<<<cdiv((long long)GATES * HIDQ, 256), 256, 0, stream>>>(Wh, WhT, HIDQ, GATES, HIDQ, GATES);
  transpose_cast_kernel<<<cdiv((long long)HIDQ * HIDQ, 256), 256, 0, stream>>>(Wq, WqT, HIDQ, HIDQ, HIDQ, HIDQ);
  transpose_cast_kernel<<<cdiv((long long)HIDQ * HIDQ, 256), 256, 0, stream>>>(Wk, WkT, HIDQ, HIDQ, HIDQ, HIDQ);
  transpose_cast_kernel<<<cdiv((long long)HIDQ * HIDQ, 256), 256, 0, stream>>>(Wv, WvT, HIDQ, HIDQ, HIDQ, HIDQ);
  transpose_cast_kernel<<<cdiv((long long)HIDQ * HIDQ, 256), 256, 0, stream>>>(Wo, WoT, HIDQ, HIDQ, HIDQ, HIDQ);
  transpose_cast_kernel<<<cdiv((long long)D1PAD * LQ * HIDQ, 256), 256, 0, stream>>>(W1, W1T, LQ * HIDQ, D1Q, LQ * HIDQ, D1PAD);
  pad_bias_kernel<<<1, D1PAD, 0, stream>>>(b1, b1p, D1Q, D1PAD);

  // ---- embedding gather + cast ----
  embed_cast_kernel<<<cdiv((long long)BQ * LQ * EPAD, 256), 256, 0, stream>>>(x, embed, XembF16);

  // ---- time-parallel input GEMM: gatesx = Xemb @ Wi + b_lstm ----
  run_gemm(stream, 1, false, XembF16, WiT, b_lstm, gatesx,
           BQ * LQ, GATES, EPAD, EPAD, EPAD, GATES, 0, 0, 0, 1, 1.0f);

  // ---- persistent recurrent LSTM (single launch, 128 steps) ----
  init_state_kernel<<<cdiv(BQ * HIDQ, 256), 256, 0, stream>>>(hPing, hPong, cSt, barCnt);
  lstm_persistent_kernel<<<NBLK_LSTM, 256, 0, stream>>>(gatesx, WhT, hPing, hPong, cSt, hseq, barCnt);

  // ---- attention projections ----
  run_gemm(stream, 1, false, hseq, WqT, bq, qbuf,
           BQ * LQ, HIDQ, HIDQ, HIDQ, HIDQ, HIDQ, 0, 0, 0, 1, 1.0f);
  run_gemm(stream, 1, false, hseq, WkT, bk, kbuf,
           BQ * LQ, HIDQ, HIDQ, HIDQ, HIDQ, HIDQ, 0, 0, 0, 1, 1.0f);
  run_gemm(stream, 2, false, hseq, WvT, bv, vT,
           BQ * LQ, HIDQ, HIDQ, HIDQ, HIDQ, 0, 0, 0, 0, 1, 1.0f);

  // ---- scores = Q K^T / 16, batched over B ----
  run_gemm(stream, 0, false, qbuf, kbuf, nullptr, scores,
           LQ, LQ, HIDQ, HIDQ, HIDQ, LQ,
           (long long)LQ * HIDQ, (long long)LQ * HIDQ, (long long)LQ * LQ, BQ, 0.0625f);

  attn_softmax_kernel<<<cdiv((long long)BQ * LQ, 8), 256, 0, stream>>>(scores, mask, attn);

  // ---- ctx = attn @ V, batched ----
  run_gemm(stream, 1, false, attn, vT, nullptr, ctx,
           LQ, HIDQ, LQ, LQ, LQ, HIDQ,
           (long long)LQ * LQ, (long long)HIDQ * LQ, (long long)LQ * HIDQ, BQ, 1.0f);

  // ---- output projection -> flat [B, L*HID] ----
  run_gemm(stream, 1, false, ctx, WoT, bo, flat,
           BQ * LQ, HIDQ, HIDQ, HIDQ, HIDQ, HIDQ, 0, 0, 0, 1, 1.0f);

  // ---- MLP layer 1 ----
  run_gemm(stream, 0, true, flat, W1T, b1p, relu1,
           BQ, D1PAD, LQ * HIDQ, LQ * HIDQ, LQ * HIDQ, D1PAD, 0, 0, 0, 1, 1.0f);

  // ---- MLP layer 2 + softmax ----
  mlp2_softmax_kernel<<<1, BQ, 0, stream>>>(relu1, W2, b2, (float*)d_out);
}
